// GINBase_11948599018375
// MI455X (gfx1250) — compile-verified, matrix operand order checked by reading, not run on hardware
//
#include <hip/hip_runtime.h>
#include <hip/hip_bf16.h>

#define NNODES 100000
#define HF 64
#define H2F 128
#define NGRAPHS 512
#define NCLASSES 10

typedef __attribute__((ext_vector_type(16))) __bf16 v16bf;
typedef __attribute__((ext_vector_type(8)))  __bf16 v8bf;
typedef __attribute__((ext_vector_type(8)))  float  v8f;

// ---------- helpers ----------
__device__ __forceinline__ unsigned enc_f32(float x) {
    unsigned b = __float_as_uint(x);
    return (b & 0x80000000u) ? ~b : (b | 0x80000000u);
}
__device__ __forceinline__ float dec_f32(unsigned u) {
    unsigned b = (u & 0x80000000u) ? (u & 0x7FFFFFFFu) : ~u;
    return __uint_as_float(b);
}

// Load one 16x32 bf16 WMMA fragment from an LDS tile stored row-major
// (row stride = ld bf16 elements). Works for A (row = M) and for B when the
// weight matrix is stored transposed [N][K] (row = N).
// Documented 16-bit A layout: lane half = lane>>4, VGPRs 0..3 hold
// K = k0 + half*8 .. +7 (contiguous -> one b128), VGPRs 4..7 hold
// K = k0 + 16 + half*8 .. +7 (second b128).
__device__ __forceinline__ v16bf load_frag(const __bf16* t, int ld, int row, int k0, int half) {
    const __bf16* p = t + row * ld + k0 + half * 8;
    union { v16bf v; v8bf h[2]; } u;
    u.h[0] = *(const v8bf*)(p);
    u.h[1] = *(const v8bf*)(p + 16);
    return u.v;
}

// ---------- kernels ----------

// zero a float buffer
__global__ void k_zero_f32(float* __restrict__ p, int n) {
    int t = blockIdx.x * blockDim.x + threadIdx.x;
    if (t < n) p[t] = 0.0f;
}

// init pooling buffers: sums=0, maxenc=enc(-inf)=0x007FFFFF, cnts=0
__global__ void k_init_pool(float* __restrict__ sums, unsigned* __restrict__ maxenc,
                            float* __restrict__ cnts, int n) {
    int t = blockIdx.x * blockDim.x + threadIdx.x;
    if (t < n) { sums[t] = 0.0f; maxenc[t] = 0x007FFFFFu; }
    if (t < NGRAPHS) cnts[t] = 0.0f;
}

// convert + transpose weights to bf16: WaT[n][k] (128x64), WbT[n][k] (64x128)
__global__ void k_prep_weights(const float* __restrict__ Wa, const float* __restrict__ Wb,
                               __bf16* __restrict__ WaT, __bf16* __restrict__ WbT) {
    int t = blockIdx.x * blockDim.x + threadIdx.x;
    if (t < HF * H2F) {
        { int n = t / HF,  k = t % HF;  WaT[t] = (__bf16)Wa[k * H2F + n]; }
        { int n = t / H2F, k = t % H2F; WbT[t] = (__bf16)Wb[k * HF  + n]; }
    }
}

// h0 = x @ W_emb + b_emb   ([N,7] @ [7,64])
__global__ void k_embed(const float* __restrict__ x, const float* __restrict__ W,
                        const float* __restrict__ b, float* __restrict__ h, int n_nodes) {
    int t = blockIdx.x * blockDim.x + threadIdx.x;
    if (t >= n_nodes * HF) return;
    int n = t >> 6, f = t & 63;
    __builtin_prefetch(x + n * 7, 0, 3);
    float acc = b[f];
#pragma unroll
    for (int k = 0; k < 7; ++k) acc += x[n * 7 + k] * W[k * HF + f];
    h[t] = acc;
}

// agg[dst] += h[src] : one thread per (edge, 4-float chunk). agg fits in L2.
__global__ void k_scatter(const float* __restrict__ h, const int* __restrict__ src,
                          const int* __restrict__ dst, float* __restrict__ agg, int n_edges) {
    int t = blockIdx.x * blockDim.x + threadIdx.x;
    int e = t >> 4;
    if (e >= n_edges) return;
    int c = (t & 15) << 2;
    int s = src[e], d = dst[e];
    const float4 v = *(const float4*)(h + s * HF + c);
    float* o = agg + d * HF + c;
    atomicAdd(o + 0, v.x);
    atomicAdd(o + 1, v.y);
    atomicAdd(o + 2, v.z);
    atomicAdd(o + 3, v.w);
}

// Fused GIN node MLP: h_out = elu( relu( ((1+eps)h + agg) @ Wa + ba ) @ Wb + bb )
// One wave per 16-node tile; bf16 WMMA 16x16x32 with f32 accumulation.
__global__ __launch_bounds__(128) void k_gin_mlp(
    const float* __restrict__ h_in, const float* __restrict__ agg,
    const float* __restrict__ eps_p,
    const __bf16* __restrict__ WaT, const float* __restrict__ ba,
    const __bf16* __restrict__ WbT, const float* __restrict__ bb,
    float* __restrict__ h_out, int n_tiles)
{
    __shared__ __bf16 sWaT[H2F * HF];      // [n=128][k=64]   16 KB
    __shared__ __bf16 sWbT[HF * H2F];      // [n=64][k=128]   16 KB
    __shared__ __bf16 sA[4][16 * HF];      // per-wave input tile   8 KB
    __shared__ __bf16 sM[4][16 * H2F];     // per-wave mid tile    16 KB

    const int tid = threadIdx.x;

    // cooperative weight staging (32-bit copies)
    {
        const unsigned* sa = (const unsigned*)WaT;  unsigned* da = (unsigned*)sWaT;
        const unsigned* sb = (const unsigned*)WbT;  unsigned* db = (unsigned*)sWbT;
        for (int i = tid; i < (H2F * HF) / 2; i += 128) da[i] = sa[i];
        for (int i = tid; i < (HF * H2F) / 2; i += 128) db[i] = sb[i];
    }
    __syncthreads();

    const int wave = tid >> 5;
    const int lane = tid & 31;
    const int tile = blockIdx.x * 4 + wave;
    if (tile >= n_tiles) return;
    const int node0 = tile * 16;
    const int half  = lane >> 4;
    const int ln    = lane & 15;

    const float scale = 1.0f + eps_p[0];
    __bf16* A = sA[wave];
    __bf16* M = sM[wave];

    // build A tile: t = (1+eps)*h + agg, converted to bf16 (16x64)
#pragma unroll
    for (int i = 0; i < 32; ++i) {
        int e = i * 32 + lane;
        int r = e >> 6, c = e & 63;
        int gi = (node0 + r) * HF + c;
        A[e] = (__bf16)(scale * h_in[gi] + agg[gi]);
    }
    __builtin_amdgcn_wave_barrier();

    // GEMM1: [16x64] @ [64x128] -> relu -> bf16 mid tile
#pragma unroll
    for (int nt = 0; nt < 8; ++nt) {
        float bv = ba[nt * 16 + ln];
        v8f acc;
#pragma unroll
        for (int r = 0; r < 8; ++r) acc[r] = bv;
#pragma unroll
        for (int kt = 0; kt < 2; ++kt) {
            v16bf af = load_frag(A,    HF, ln,           kt * 32, half);
            v16bf bf = load_frag(sWaT, HF, nt * 16 + ln, kt * 32, half);
            acc = __builtin_amdgcn_wmma_f32_16x16x32_bf16(
                      false, af, false, bf, (short)0, acc, false, false);
        }
#pragma unroll
        for (int r = 0; r < 8; ++r) {
            int m = r + 8 * half;                       // D row per documented C/D layout
            float f = acc[r] > 0.0f ? acc[r] : 0.0f;    // relu
            M[m * H2F + nt * 16 + ln] = (__bf16)f;
        }
    }
    __builtin_amdgcn_wave_barrier();

    // GEMM2: [16x128] @ [128x64] -> elu -> f32 out
#pragma unroll
    for (int nt = 0; nt < 4; ++nt) {
        float bv = bb[nt * 16 + ln];
        v8f acc;
#pragma unroll
        for (int r = 0; r < 8; ++r) acc[r] = bv;
#pragma unroll
        for (int kt = 0; kt < 4; ++kt) {
            v16bf af = load_frag(M,    H2F, ln,           kt * 32, half);
            v16bf bf = load_frag(sWbT, H2F, nt * 16 + ln, kt * 32, half);
            acc = __builtin_amdgcn_wmma_f32_16x16x32_bf16(
                      false, af, false, bf, (short)0, acc, false, false);
        }
#pragma unroll
        for (int r = 0; r < 8; ++r) {
            int m = r + 8 * half;
            float v = acc[r];
            float o = v > 0.0f ? v : (__expf(v) - 1.0f);  // elu
            h_out[(node0 + m) * HF + nt * 16 + ln] = o;
        }
    }
}

// per-graph pooling: sums += h, maxenc = max(enc(h)), cnts += 1
__global__ void k_pool(const float* __restrict__ h, const int* __restrict__ batch,
                       float* __restrict__ sums, unsigned* __restrict__ maxenc,
                       float* __restrict__ cnts, int n_nodes) {
    int t = blockIdx.x * blockDim.x + threadIdx.x;
    if (t >= n_nodes * HF) return;
    int n = t >> 6, f = t & 63;
    int g = batch[n];
    float v = h[t];
    atomicAdd(&sums[g * HF + f], v);
    atomicMax(&maxenc[g * HF + f], enc_f32(v));
    if (f == 0) atomicAdd(&cnts[g], 1.0f);
}

// classifier: g=[mean,max] (128) -> relu(g@Wc1+bc1) (64) -> @Wc2+bc2 (10)
__global__ __launch_bounds__(64) void k_classifier(
    const float* __restrict__ sums, const unsigned* __restrict__ maxenc,
    const float* __restrict__ cnts,
    const float* __restrict__ Wc1, const float* __restrict__ bc1,
    const float* __restrict__ Wc2, const float* __restrict__ bc2,
    float* __restrict__ out)
{
    __shared__ float gf[2 * HF];
    __shared__ float hid[HF];
    int g = blockIdx.x, t = threadIdx.x;
    float inv = 1.0f / fmaxf(cnts[g], 1.0f);
    gf[t]      = sums[g * HF + t] * inv;
    gf[HF + t] = dec_f32(maxenc[g * HF + t]);
    __syncthreads();
    float a = bc1[t];
#pragma unroll 8
    for (int k = 0; k < 2 * HF; ++k) a += gf[k] * Wc1[k * HF + t];
    hid[t] = fmaxf(a, 0.0f);
    __syncthreads();
    if (t < NCLASSES) {
        float o = bc2[t];
#pragma unroll 8
        for (int k = 0; k < HF; ++k) o += hid[k] * Wc2[k * NCLASSES + t];
        out[g * NCLASSES + t] = o;
    }
}

// ---------- launch ----------
extern "C" void kernel_launch(void* const* d_in, const int* in_sizes, int n_in,
                              void* d_out, int out_size, void* d_ws, size_t ws_size,
                              hipStream_t stream) {
    const float* x      = (const float*)d_in[0];
    const int*   eidx   = (const int*)d_in[1];
    const int*   batch  = (const int*)d_in[2];
    const float* W_emb  = (const float*)d_in[3];
    const float* b_emb  = (const float*)d_in[4];
    const float* epsv[3] = { (const float*)d_in[5], (const float*)d_in[10], (const float*)d_in[15] };
    const float* Wa[3]   = { (const float*)d_in[6], (const float*)d_in[11], (const float*)d_in[16] };
    const float* ba[3]   = { (const float*)d_in[7], (const float*)d_in[12], (const float*)d_in[17] };
    const float* Wb[3]   = { (const float*)d_in[8], (const float*)d_in[13], (const float*)d_in[18] };
    const float* bb[3]   = { (const float*)d_in[9], (const float*)d_in[14], (const float*)d_in[19] };
    const float* Wc1 = (const float*)d_in[20];
    const float* bc1 = (const float*)d_in[21];
    const float* Wc2 = (const float*)d_in[22];
    const float* bc2 = (const float*)d_in[23];

    const int n_edges = in_sizes[1] / 2;
    const int* src = eidx;
    const int* dst = eidx + n_edges;

    // workspace carve (256-byte aligned chunks)
    char* p = (char*)d_ws;
    auto take = [&](size_t bytes) { char* q = p; p += (bytes + 255) & ~(size_t)255; return q; };
    float*    hA     = (float*)take((size_t)NNODES * HF * sizeof(float));
    float*    hB     = (float*)take((size_t)NNODES * HF * sizeof(float));
    float*    agg    = (float*)take((size_t)NNODES * HF * sizeof(float));
    __bf16*   WaT[3]; __bf16* WbT[3];
    for (int l = 0; l < 3; ++l) {
        WaT[l] = (__bf16*)take((size_t)HF * H2F * sizeof(__bf16));
        WbT[l] = (__bf16*)take((size_t)HF * H2F * sizeof(__bf16));
    }
    float*    sums   = (float*)take((size_t)NGRAPHS * HF * sizeof(float));
    unsigned* maxenc = (unsigned*)take((size_t)NGRAPHS * HF * sizeof(unsigned));
    float*    cnts   = (float*)take((size_t)NGRAPHS * sizeof(float));

    const int nfeat_total = NNODES * HF;
    const int blk = 256;
    const int grid_feat = (nfeat_total + blk - 1) / blk;
    const int n_tiles = NNODES / 16;                 // 6250, exact
    const int grid_mlp = (n_tiles + 3) / 4;          // 4 waves (tiles) per block

    // weight prep (every call; no caching allowed)
    for (int l = 0; l < 3; ++l)
        k_prep_weights<<<(HF * H2F + blk - 1) / blk, blk, 0, stream>>>(Wa[l], Wb[l], WaT[l], WbT[l]);

    // embedding
    k_embed<<<grid_feat, blk, 0, stream>>>(x, W_emb, b_emb, hA, NNODES);

    float* cur = hA;
    float* nxt = hB;
    for (int l = 0; l < 3; ++l) {
        k_zero_f32<<<grid_feat, blk, 0, stream>>>(agg, nfeat_total);
        k_scatter<<<((n_edges * 16) + blk - 1) / blk, blk, 0, stream>>>(cur, src, dst, agg, n_edges);
        k_gin_mlp<<<grid_mlp, 128, 0, stream>>>(cur, agg, epsv[l], WaT[l], ba[l], WbT[l], bb[l], nxt, n_tiles);
        float* t = cur; cur = nxt; nxt = t;
    }

    // pooling + classifier
    k_init_pool<<<(NGRAPHS * HF + blk - 1) / blk, blk, 0, stream>>>(sums, maxenc, cnts, NGRAPHS * HF);
    k_pool<<<grid_feat, blk, 0, stream>>>(cur, batch, sums, maxenc, cnts, NNODES);
    k_classifier<<<NGRAPHS, 64, 0, stream>>>(sums, maxenc, cnts, Wc1, bc1, Wc2, bc2, (float*)d_out);
}
